// BayesianCTC_36266703847809
// MI455X (gfx1250) — matrix operand order repaired and checked by reading, compiled, not run on hardware
//
#include <hip/hip_runtime.h>
#include <hip/hip_bf16.h>
#include <cstdint>

// ---------------------------------------------------------------------------
// Bayesian CTC loss for MI455X (gfx1250, wave32).
// Pipeline:
//   1) fp32 -> f16 conversion of hs_pad and W
//   2) WMMA GEMM (v_wmma_f32_16x16x32_f16): logits = hs @ W^T + b
//      - A panel (32x512 f16) staged into LDS via Tensor Data Mover
//        (tensor_load_to_lds + s_wait_tensorcnt), read back with ds_load_b128
//   3) fused log_softmax (f32, as in reference) + emission table em (f64)
//   4) alpha forward scan  (f64, LDS ping-pong, 1 block / batch)
//   5) beta backward scan  (f64)
//   6) per-(b,u) LSE over T of alpha+beta'+risk  (block reductions)
//   7) finalize: last finite u, mean of -loss  -> d_out[0] (float)
// ---------------------------------------------------------------------------

typedef __attribute__((ext_vector_type(16))) _Float16     v16h;
typedef __attribute__((ext_vector_type(8)))  float        v8f;
typedef __attribute__((ext_vector_type(4)))  unsigned int v4u;
typedef __attribute__((ext_vector_type(8)))  int          v8i;
typedef __attribute__((ext_vector_type(4)))  int          v4i;

constexpr int Bn = 16, Tn = 1600, En = 512, On = 2048, Un = 200;
constexpr int Sn = 2 * Un + 1;     // 401
constexpr int BT = Bn * Tn;        // 25600
constexpr double RISK = 0.1;

#define DEVINL __device__ __forceinline__

union F16Frag { v16h v; uint4 q[2]; };

// ---------------------------------------------------------------------------
// fp32 -> f16 conversion (grid-stride)
// ---------------------------------------------------------------------------
__global__ void k_cvt_f16(const float* __restrict__ src,
                          _Float16* __restrict__ dst, size_t n) {
  size_t i = (size_t)blockIdx.x * blockDim.x + threadIdx.x;
  const size_t stride = (size_t)gridDim.x * blockDim.x;
  for (; i < n; i += stride) dst[i] = (_Float16)src[i];
}

// ---------------------------------------------------------------------------
// WMMA GEMM: logits(BT,On) = A(BT,En) * W(On,En)^T + bias
// Block: 128 threads = 4 waves; block tile = 32 rows x 256 cols.
// Each wave computes 32x64 (2 M-tiles x 4 N-tiles = 8 accumulators).
// A panel (32x512 f16 = 32 KB) staged into LDS with the TDM, A-frags read via
// ds_load_b128; B-frags streamed from global (W stays L2-resident at 2 MB).
// Grid: (On/256, BT/32).
// ---------------------------------------------------------------------------
__global__ __launch_bounds__(128) void k_gemm_logits(
    const _Float16* __restrict__ A,
    const _Float16* __restrict__ Wt,
    const float* __restrict__ bias,
    float* __restrict__ logits) {
  __shared__ _Float16 shA[32 * En];   // 32 KB

  const int lane = threadIdx.x & 31;
  const int wave = threadIdx.x >> 5;
  const int row0 = blockIdx.y << 5;                 // 32 rows per block
  const int col0 = (blockIdx.x << 8) + (wave << 6); // 64 cols per wave
  const int r  = lane & 15;
  const int hi = lane >> 4;

#if __has_builtin(__builtin_amdgcn_tensor_load_to_lds) && \
    __has_builtin(__builtin_amdgcn_s_wait_tensorcnt)
  // --- Tensor Data Mover: DMA the 32x512 f16 A panel into LDS (wave 0) ---
  if (threadIdx.x < 32) {
    // Flat->LDS aperture maps addr[31:0] to the LDS byte offset.
    const unsigned lds_addr =
        (unsigned)(uintptr_t)(&shA[0]);
    const unsigned long long ga =
        (unsigned long long)(uintptr_t)(A + (size_t)row0 * En);
    // D# group 0: count=1 | lds_addr | global_addr[56:0] | type=2
    v4u g0;
    g0[0] = 1u;                                        // count=1, user mode
    g0[1] = lds_addr;                                  // lds_addr (bytes)
    g0[2] = (unsigned)(ga & 0xFFFFFFFFu);              // global_addr[31:0]
    g0[3] = (unsigned)((ga >> 32) & 0x01FFFFFFu)       // global_addr[56:32]
            | (2u << 30);                              // type = 2 ("image")
    // D# group 1: data_size=1 (2B) | tensor 512x25600 | tile 512x32 | stride 512
    v8i g1;
    g1[0] = (int)(1u << 16);                           // data_size=2 bytes
    g1[1] = (int)((unsigned)(En & 0xFFFF) << 16);      // tensor_dim0[15:0]
    g1[2] = (int)((unsigned)(BT & 0xFFFF) << 16);      // tensor_dim1[15:0]
    g1[3] = (int)((unsigned)(En & 0xFFFF) << 16);      // tile_dim0 = 512
    g1[4] = 32;                                        // tile_dim1 = 32
    g1[5] = En;                                        // tensor_dim0_stride
    g1[6] = 0;
    g1[7] = 0;
    v4i g2 = {0, 0, 0, 0};
    v4i g3 = {0, 0, 0, 0};
    v8i g4 = {0, 0, 0, 0, 0, 0, 0, 0};                 // trailing group (6-arg form)
    __builtin_amdgcn_tensor_load_to_lds(g0, g1, g2, g3, g4, /*cpol=*/0);
    __builtin_amdgcn_s_wait_tensorcnt(0);
  }
#else
  // Fallback: cooperative 16B copies (uint4) of the 32x512 panel.
  {
    const uint4* src = (const uint4*)(A + (size_t)row0 * En);
    uint4* dst = (uint4*)(&shA[0]);
    const int nvec = 32 * En / 8;                      // 2048 uint4s
    for (int i = threadIdx.x; i < nvec; i += 128) dst[i] = src[i];
  }
#endif
  __syncthreads();

  v8f acc[8] = {};                                     // [m*4 + n]
  for (int k0 = 0; k0 < En; k0 += 32) {
    F16Frag af[2];
#pragma unroll
    for (int m = 0; m < 2; ++m) {
      // A-frag (16-bit A 16x32 layout): lane<16 holds K chunks [0..7],[16..23]
      // of row r; lane>=16 holds [8..15],[24..31].
      const _Float16* ap = &shA[(size_t)(m * 16 + r) * En + k0 + hi * 8];
      af[m].q[0] = *(const uint4*)(ap);                // ds_load_b128
      af[m].q[1] = *(const uint4*)(ap + 16);
    }
#pragma unroll
    for (int n = 0; n < 4; ++n) {
      // B-frag (16-bit B 32x16): lane&15 = column, lane<16: K [0..7],[16..23],
      // lane>=16: K [8..15],[24..31] (matches A-frag K order).
      F16Frag bf;
      const _Float16* bp =
          Wt + (size_t)(col0 + n * 16 + r) * En + k0 + hi * 8;
      bf.q[0] = *(const uint4*)(bp);
      bf.q[1] = *(const uint4*)(bp + 16);
      if (n == 0 && k0 + 32 < En)
        __builtin_prefetch(bp + 32, 0, 1);             // global_prefetch_b8
      acc[n] = __builtin_amdgcn_wmma_f32_16x16x32_f16(
          false, af[0].v, false, bf.v, (short)0, acc[n], false, false);
      acc[4 + n] = __builtin_amdgcn_wmma_f32_16x16x32_f16(
          false, af[1].v, false, bf.v, (short)0, acc[4 + n], false, false);
    }
  }

  // C layout: lane<16 -> N=lane, rows i+0; lane>=16 -> N=lane-16, rows i+8.
#pragma unroll
  for (int m = 0; m < 2; ++m) {
#pragma unroll
    for (int n = 0; n < 4; ++n) {
      const int ccol = col0 + n * 16 + r;
      const float bv = bias[ccol];
#pragma unroll
      for (int i = 0; i < 8; ++i) {
        const int rr = row0 + m * 16 + hi * 8 + i;
        logits[(size_t)rr * On + ccol] = acc[m * 4 + n][i] + bv;
      }
    }
  }
}

// ---------------------------------------------------------------------------
// Fused log_softmax (f32, matching reference) + emission table build.
// em[b,t,s]: even s -> lp_blank; odd s=2u+1 -> lp_label[u]; s=2U -> lp_blank.
// ---------------------------------------------------------------------------
__global__ __launch_bounds__(256) void k_softmax_em(
    const float* __restrict__ logits,
    const long long* __restrict__ ys,
    double* __restrict__ em) {
  const int row = blockIdx.x;           // b*Tn + t
  const int b = row / Tn;
  const int tid = threadIdx.x;
  const float* lg = logits + (size_t)row * On;
  __shared__ float red[256];

  float m = -INFINITY;
  for (int j = tid; j < On; j += 256) m = fmaxf(m, lg[j]);
  red[tid] = m; __syncthreads();
  for (int off = 128; off > 0; off >>= 1) {
    if (tid < off) red[tid] = fmaxf(red[tid], red[tid + off]);
    __syncthreads();
  }
  const float mx = red[0];
  __syncthreads();

  float sum = 0.f;
  for (int j = tid; j < On; j += 256) sum += expf(lg[j] - mx);
  red[tid] = sum; __syncthreads();
  for (int off = 128; off > 0; off >>= 1) {
    if (tid < off) red[tid] += red[tid + off];
    __syncthreads();
  }
  const float logZ = mx + logf(red[0]);

  const float lpb = lg[0] - logZ;
  double* emr = em + (size_t)row * Sn;
  for (int s = tid; s < Sn; s += 256) {
    double v;
    if (s & 1) {
      const long long y = ys[(size_t)b * Un + (s >> 1)];
      const int cls = (y < 0) ? 0 : (int)y;
      v = (double)(lg[cls] - logZ);
    } else {
      v = (double)lpb;
    }
    emr[s] = v;
  }
}

// ---------------------------------------------------------------------------
// f64 helpers matching the reference's semantics
// ---------------------------------------------------------------------------
DEVINL double lse3(double a, double b, double c) {
  const double m = fmax(a, fmax(b, c));
  if (isinf(m)) return -INFINITY;       // all -inf case of _safe_lse0
  return m + log(exp(a - m) + exp(b - m) + exp(c - m));
}

DEVINL double log_sub_exp(double a, double b) {
  if (!isinf(a) && !isinf(b)) {
    const double tmp = b + log(exp(a - b) - 1.0);   // NaN if a<b (propagates)
    if (isinf(tmp)) return -2001.0 + log(exp(1.0) - 1.0);  // reference fallback
    return tmp;
  }
  if (!isinf(a) && isinf(b)) return a;
  return -INFINITY;
}

// ---------------------------------------------------------------------------
// Alpha forward scan: one block per batch element, parallel over S states.
// ---------------------------------------------------------------------------
__global__ __launch_bounds__(512) void k_alpha(
    const double* __restrict__ em,
    const long long* __restrict__ ys,
    double* __restrict__ alpha_odd) {   // (B,T,U)
  const int b = blockIdx.x;
  const int tid = threadIdx.x;
  __shared__ double buf[2][Sn];
  __shared__ unsigned char allow[Sn];

  for (int s = tid; s < Sn; s += blockDim.x) {
    bool a = false;
    if ((s & 1) && s >= 3) {            // odd s=2u+1, u>=1
      const int u = s >> 1;
      const long long y0 = ys[(size_t)b * Un + u - 1];
      const long long y1 = ys[(size_t)b * Un + u];
      const int c0 = (y0 < 0) ? 0 : (int)y0;
      const int c1 = (y1 < 0) ? 0 : (int)y1;
      a = (c0 != c1);
    }
    allow[s] = a ? 1 : 0;
  }

  const double* emb = em + (size_t)b * Tn * Sn;
  for (int s = tid; s < Sn; s += blockDim.x) {
    const double v = (s == 0) ? emb[0] : ((s == 1) ? emb[1] : -INFINITY);
    buf[0][s] = v;
    if (s & 1) alpha_odd[((size_t)b * Tn + 0) * Un + (s >> 1)] = v;
  }
  __syncthreads();

  int p = 0;
  for (int t = 1; t < Tn; ++t) {
    const double* e = emb + (size_t)t * Sn;
    const double* pv = buf[p];
    double* nb = buf[p ^ 1];
    for (int s = tid; s < Sn; s += blockDim.x) {
      const double x0 = pv[s];
      const double x1 = (s >= 1) ? pv[s - 1] : -INFINITY;
      const double x2 = (s >= 2 && allow[s]) ? pv[s - 2] : -INFINITY;
      const double nv = e[s] + lse3(x0, x1, x2);
      nb[s] = nv;
      if (s & 1) alpha_odd[((size_t)b * Tn + t) * Un + (s >> 1)] = nv;
    }
    p ^= 1;
    __syncthreads();
  }
}

// ---------------------------------------------------------------------------
// Beta backward scan (matches reference: e_next = em[min(t+1,T-1)],
// fin replacement at t == hlen-1).
// ---------------------------------------------------------------------------
__global__ __launch_bounds__(512) void k_beta(
    const double* __restrict__ em,
    const long long* __restrict__ ys,
    const long long* __restrict__ hlens,
    double* __restrict__ beta_odd) {    // (B,T,U)
  const int b = blockIdx.x;
  const int tid = threadIdx.x;
  __shared__ double buf[2][Sn];
  __shared__ unsigned char allow_fwd[Sn];
  __shared__ int sh_olen;

  if (tid == 0) {
    int o = 0;
    for (int u = 0; u < Un; ++u) o += (ys[(size_t)b * Un + u] >= 0) ? 1 : 0;
    sh_olen = o;
  }
  for (int s = tid; s < Sn; s += blockDim.x) {
    bool a = false;
    const int s2 = s + 2;               // allow_fwd[s] = allow[s+2]
    if (s2 < Sn && (s2 & 1) && s2 >= 3) {
      const int u = s2 >> 1;
      const long long y0 = ys[(size_t)b * Un + u - 1];
      const long long y1 = ys[(size_t)b * Un + u];
      const int c0 = (y0 < 0) ? 0 : (int)y0;
      const int c1 = (y1 < 0) ? 0 : (int)y1;
      a = (c0 != c1);
    }
    allow_fwd[s] = a ? 1 : 0;
    buf[0][s] = -INFINITY;
  }
  __syncthreads();

  const long long hlen = hlens[b];
  const int olen = sh_olen;
  const double* emb = em + (size_t)b * Tn * Sn;

  int p = 0;
  for (int t = Tn - 1; t >= 0; --t) {
    const int tn = (t + 1 < Tn) ? (t + 1) : (Tn - 1);
    const double* e = emb + (size_t)tn * Sn;
    const double* pv = buf[p];
    double* nb = buf[p ^ 1];
    for (int s = tid; s < Sn; s += blockDim.x) {
      const double g0 = e[s] + pv[s];
      const double g1 = (s + 1 < Sn) ? (e[s + 1] + pv[s + 1]) : -INFINITY;
      const double g2 = (s + 2 < Sn && allow_fwd[s]) ? (e[s + 2] + pv[s + 2])
                                                     : -INFINITY;
      double nv = lse3(g0, g1, g2);
      if ((long long)t == hlen - 1)
        nv = (s == 2 * olen || s == 2 * olen - 1) ? 0.0 : -INFINITY;
      nb[s] = nv;
      if (s & 1) beta_odd[((size_t)b * Tn + t) * Un + (s >> 1)] = nv;
    }
    p ^= 1;
    __syncthreads();
  }
}

// ---------------------------------------------------------------------------
// loss_state(b,u,t) = alpha_u + beta' + risk  (NaN -> -inf), recomputed in
// both reduction passes (cheap relative to a spill buffer).
// ---------------------------------------------------------------------------
DEVINL double loss_state_val(const double* __restrict__ ao,
                             const double* __restrict__ bo,
                             const double* __restrict__ em,
                             int b, int u, int t, int olen, long long hlen) {
  const bool valid = (u < olen) && ((long long)t < hlen);
  const size_t base = ((size_t)b * Tn + t) * Un + u;
  const double au = valid ? ao[base] : -INFINITY;
  const double bu = valid ? bo[base] : -INFINITY;
  double bp;
  if (t < Tn - 1) {
    const bool v1 = (u < olen) && ((long long)(t + 1) < hlen);
    const double bu1 = v1 ? bo[base + Un] : -INFINITY;
    const double p1  = v1 ? em[((size_t)b * Tn + (t + 1)) * Sn + (2 * u + 1)]
                          : -INFINITY;
    bp = log_sub_exp(bu, bu1 + p1);
  } else {
    bp = bu;
  }
  const double risk = ((double)(t + 1)) / (double)hlen * RISK;
  double ls = au + bp + risk;
  if (isnan(ls)) ls = -INFINITY;
  return ls;
}

__global__ __launch_bounds__(256) void k_loss_u(
    const double* __restrict__ alpha_odd,
    const double* __restrict__ beta_odd,
    const double* __restrict__ em,
    const long long* __restrict__ ys,
    const long long* __restrict__ hlens,
    double* __restrict__ loss_u) {      // (B,U)
  const int b = blockIdx.x;
  const int tid = threadIdx.x;
  __shared__ double red[256];
  const long long hlen = hlens[b];
  int olen = 0;
  for (int u = 0; u < Un; ++u) olen += (ys[(size_t)b * Un + u] >= 0) ? 1 : 0;

  for (int u = 0; u < Un; ++u) {
    double mx = -INFINITY;
    for (int t = tid; t < Tn; t += 256)
      mx = fmax(mx, loss_state_val(alpha_odd, beta_odd, em, b, u, t, olen, hlen));
    red[tid] = mx; __syncthreads();
    for (int off = 128; off > 0; off >>= 1) {
      if (tid < off) red[tid] = fmax(red[tid], red[tid + off]);
      __syncthreads();
    }
    const double m = red[0];
    const double ms = isinf(m) ? 0.0 : m;
    __syncthreads();

    double s = 0.0;
    for (int t = tid; t < Tn; t += 256)
      s += exp(loss_state_val(alpha_odd, beta_odd, em, b, u, t, olen, hlen) - ms);
    red[tid] = s; __syncthreads();
    for (int off = 128; off > 0; off >>= 1) {
      if (tid < off) red[tid] += red[tid + off];
      __syncthreads();
    }
    if (tid == 0) {
      const double ss = red[0];
      loss_u[(size_t)b * Un + u] = (ss == 0.0) ? -INFINITY : (ms + log(ss));
    }
    __syncthreads();
  }
}

// ---------------------------------------------------------------------------
// Finalize: pick last finite u per batch (wrap negative index like JAX),
// zero when hlen < olen, mean of -loss.
// ---------------------------------------------------------------------------
__global__ void k_finalize(const double* __restrict__ loss_u,
                           const long long* __restrict__ ys,
                           const long long* __restrict__ hlens,
                           float* __restrict__ out) {
  if (threadIdx.x == 0 && blockIdx.x == 0) {
    double acc = 0.0;
    for (int b = 0; b < Bn; ++b) {
      int olen = 0;
      for (int u = 0; u < Un; ++u) olen += (ys[(size_t)b * Un + u] >= 0) ? 1 : 0;
      int cnt = 0;
      for (int u = 0; u < Un; ++u)
        cnt += isinf(loss_u[(size_t)b * Un + u]) ? 0 : 1;
      int last = cnt - 1;
      last = ((last % Un) + Un) % Un;    // JAX wrap semantics for -1
      double lf = loss_u[(size_t)b * Un + last];
      if (hlens[b] < (long long)olen) lf = 0.0;
      acc += -lf;
    }
    out[0] = (float)(acc / (double)Bn);
  }
}

// ---------------------------------------------------------------------------
// Host side
// ---------------------------------------------------------------------------
static inline size_t align_up_256(size_t x) { return (x + 255) & ~(size_t)255; }

extern "C" void kernel_launch(void* const* d_in, const int* in_sizes, int n_in,
                              void* d_out, int out_size, void* d_ws, size_t ws_size,
                              hipStream_t stream) {
  (void)in_sizes; (void)n_in; (void)out_size; (void)ws_size;
  const float*     hs    = (const float*)d_in[0];     // (B,T,E) f32
  const float*     W     = (const float*)d_in[1];     // (O,E)   f32
  const float*     bias  = (const float*)d_in[2];     // (O,)    f32
  const long long* hlens = (const long long*)d_in[3]; // (B,)    i64
  const long long* ys    = (const long long*)d_in[4]; // (B,U)   i64
  // d_in[5] = ali (unused by the reference loss)
  float* out = (float*)d_out;

  char* ws = (char*)d_ws;
  size_t off = 0;
  _Float16* hs16   = (_Float16*)(ws + off); off += align_up_256((size_t)BT * En * 2);
  _Float16* w16    = (_Float16*)(ws + off); off += align_up_256((size_t)On * En * 2);
  float*    logits = (float*)(ws + off);    off += align_up_256((size_t)BT * On * 4);
  double*   em     = (double*)(ws + off);   off += align_up_256((size_t)BT * Sn * 8);
  double*   alpha  = (double*)(ws + off);   off += align_up_256((size_t)BT * Un * 8);
  double*   beta   = (double*)(ws + off);   off += align_up_256((size_t)BT * Un * 8);
  double*   lossu  = (double*)(ws + off);   off += align_up_256((size_t)Bn * Un * 8);

  k_cvt_f16<<<2048, 256, 0, stream>>>(hs, hs16, (size_t)BT * En);
  k_cvt_f16<<<512, 256, 0, stream>>>(W, w16, (size_t)On * En);
  k_gemm_logits<<<dim3(On / 256, BT / 32), 128, 0, stream>>>(hs16, w16, bias, logits);
  k_softmax_em<<<BT, 256, 0, stream>>>(logits, ys, em);
  k_alpha<<<Bn, 512, 0, stream>>>(em, ys, alpha);
  k_beta<<<Bn, 512, 0, stream>>>(em, ys, hlens, beta);
  k_loss_u<<<Bn, 256, 0, stream>>>(alpha, beta, em, ys, hlens, lossu);
  k_finalize<<<1, 32, 0, stream>>>(lossu, ys, hlens, out);
}
